// TransBlock_33277406609454
// MI455X (gfx1250) — matrix operand order repaired and checked by reading, not compiled, not run on hardware
//
#include <hip/hip_runtime.h>

// ---------------------------------------------------------------------------
// CDNA5 (gfx1250) transformer block forward.
// All GEMMs + attention via v_wmma_f32_16x16x32_bf16 (wave32 WMMA).
// All WMMA fragments are contiguous 32B LDS reads (ds_load_b128 pairs):
// weights are stored TRANSPOSED ([N][K]) by the conversion kernels, and the
// attention V tile is transposed into LDS, so no strided u16 gathers remain.
// GEMM tile staging uses global_load_async_to_lds_b128 + s_wait_asynccnt.
// ---------------------------------------------------------------------------

typedef __attribute__((ext_vector_type(16))) __bf16 v16bf;
typedef __attribute__((ext_vector_type(8)))  float  v8f;

union BF16Frag { unsigned u[8]; v16bf v; };
union F32x8    { float f[8]; v8f v; };

#if __has_builtin(__builtin_amdgcn_global_load_async_to_lds_b128)
#define USE_ASYNC_LDS 1
typedef int v4i_gcc __attribute__((vector_size(16)));
typedef __attribute__((address_space(1))) v4i_gcc* as1_v4i_p;
typedef __attribute__((address_space(3))) v4i_gcc* as3_v4i_p;
#endif

__device__ __forceinline__ void wait_async_then_barrier() {
#if defined(USE_ASYNC_LDS)
#if __has_builtin(__builtin_amdgcn_s_wait_asynccnt)
  __builtin_amdgcn_s_wait_asynccnt(0);
#else
  asm volatile("s_wait_asynccnt 0" ::: "memory");
#endif
#endif
  __syncthreads();
}

__device__ __forceinline__ unsigned short f2bf(float f) {
  unsigned x = __float_as_uint(f);
  unsigned r = (x + 0x7FFFu + ((x >> 16) & 1u)) >> 16;  // RNE
  return (unsigned short)r;
}

// A-fragment from LDS tile stored [row][K] row-major.
// WMMA 16-bit A layout: lane m = row0+(lane&15); VGPR i holds K pair at
// k = (i>>2)*16 + (lane>>4)*8 + (i&3)*2  ->  two contiguous 16B chunks.
// Requires (ld*2) % 16 == 0 and k0 multiple of 8.
__device__ __forceinline__ v16bf load_frag_a(const unsigned short* lds, int ld,
                                             int row0, int k0, int lane) {
  const int hf = lane >> 4;
  const unsigned short* base = lds + (row0 + (lane & 15)) * ld + k0 + (hf << 3);
  BF16Frag fr;
  *(uint4*)&fr.u[0] = *(const uint4*)(base);       // K = k0+hf*8   .. +7
  *(uint4*)&fr.u[4] = *(const uint4*)(base + 16);  // K = k0+16+hf*8.. +7
  return fr.v;
}

// B-fragment from LDS tile stored [N][K] (B transposed, row-major).
// WMMA 16-bit B layout: lane n = n0+(lane&15); VGPR i holds K pair at
// k = (i>>2)*16 + (lane>>4)*8 + (i&3)*2  ->  two contiguous 16B chunks.
__device__ __forceinline__ v16bf load_frag_b_nk(const unsigned short* lds, int ld,
                                                int n0, int k0, int lane) {
  const unsigned short* base =
      lds + (n0 + (lane & 15)) * ld + k0 + ((lane >> 4) << 3);
  BF16Frag fr;
  *(uint4*)&fr.u[0] = *(const uint4*)(base);
  *(uint4*)&fr.u[4] = *(const uint4*)(base + 16);
  return fr.v;
}

// ---------------------------------------------------------------------------
// bf16 GEMM: C[M,N] = A[M,K] * Bt[N,K]^T (+bias) (ReLU) (+residual).
// Block tile 64(M) x 256(N), BK=64, 256 threads = 8 waves (2x4 wave grid),
// wave tile 32x64 = eight 16x16 WMMA tiles, 16 WMMA per LDS round.
// M%64==0, N%256==0, K%64==0 (true for all uses here).
// ---------------------------------------------------------------------------
__global__ __launch_bounds__(256) void gemm_bf16_kernel(
    const unsigned short* __restrict__ A,   // [M][K] bf16
    const unsigned short* __restrict__ Bt,  // [N][K] bf16 (B transposed)
    const float* __restrict__ bias, const float* __restrict__ resid,
    float* __restrict__ outF, unsigned short* __restrict__ outB,
    int M, int N, int K, int relu) {
  __shared__ __align__(16) unsigned short As[64][80];    // 64 used + pad (160B rows)
  __shared__ __align__(16) unsigned short Bs[256][80];   // [n][k]
  const int tid = threadIdx.x, lane = tid & 31, wid = tid >> 5;
  const int bm = blockIdx.y * 64, bn = blockIdx.x * 256;
  const int wM = (wid >> 2) * 32, wN = (wid & 3) * 64;

  F32x8 acc[2][4];
#pragma unroll
  for (int i = 0; i < 2; ++i)
#pragma unroll
    for (int j = 0; j < 4; ++j)
#pragma unroll
      for (int r = 0; r < 8; ++r) acc[i][j].f[r] = 0.f;

  const int ar = tid >> 2, ac = (tid & 3) << 4;  // A tile 64x64: 16 u16/thread

  for (int k0 = 0; k0 < K; k0 += 64) {
    const unsigned short* ags = A + (size_t)(bm + ar) * K + k0 + ac;
    const unsigned short* bgs = Bt + (size_t)(bn + tid) * K + k0;  // 64 u16/thread
#if defined(USE_ASYNC_LDS)
    __builtin_amdgcn_global_load_async_to_lds_b128(
        (as1_v4i_p)(ags), (as3_v4i_p)&As[ar][ac], 0, 0);
    __builtin_amdgcn_global_load_async_to_lds_b128(
        (as1_v4i_p)(ags + 8), (as3_v4i_p)&As[ar][ac + 8], 0, 0);
#pragma unroll
    for (int j = 0; j < 8; ++j)
      __builtin_amdgcn_global_load_async_to_lds_b128(
          (as1_v4i_p)(bgs + j * 8), (as3_v4i_p)&Bs[tid][j * 8], 0, 0);
#else
    const uint4* ag = (const uint4*)ags;
    uint4* ad = (uint4*)&As[ar][ac];
    ad[0] = ag[0];
    ad[1] = ag[1];
    const uint4* bg = (const uint4*)bgs;
    uint4* bd = (uint4*)&Bs[tid][0];
#pragma unroll
    for (int j = 0; j < 8; ++j) bd[j] = bg[j];
    if (k0 + 64 < K) {
      __builtin_prefetch(ags + 64, 0, 1);
      __builtin_prefetch(bgs + 64, 0, 1);
    }
#endif
    wait_async_then_barrier();

#pragma unroll
    for (int kk = 0; kk < 64; kk += 32) {
      v16bf aF0 = load_frag_a(&As[0][0], 80, wM, kk, lane);
      v16bf aF1 = load_frag_a(&As[0][0], 80, wM + 16, kk, lane);
#pragma unroll
      for (int nt = 0; nt < 4; ++nt) {
        v16bf bF = load_frag_b_nk(&Bs[0][0], 80, wN + nt * 16, kk, lane);
        acc[0][nt].v = __builtin_amdgcn_wmma_f32_16x16x32_bf16(
            false, aF0, false, bF, (short)0, acc[0][nt].v, false, false);
        acc[1][nt].v = __builtin_amdgcn_wmma_f32_16x16x32_bf16(
            false, aF1, false, bF, (short)0, acc[1][nt].v, false, false);
      }
    }
    __syncthreads();
  }

  const int hf = lane >> 4, ln = lane & 15;
#pragma unroll
  for (int mt = 0; mt < 2; ++mt)
#pragma unroll
    for (int nt = 0; nt < 4; ++nt)
#pragma unroll
      for (int r = 0; r < 8; ++r) {
        int m = bm + wM + mt * 16 + r + 8 * hf;
        int n = bn + wN + nt * 16 + ln;
        float v = acc[mt][nt].f[r];
        if (bias) v += bias[n];
        if (relu) v = fmaxf(v, 0.f);
        size_t idx = (size_t)m * N + n;
        if (resid) v += resid[idx];
        if (outF) outF[idx] = v;
        else      outB[idx] = f2bf(v);
      }
}

// ---------------------------------------------------------------------------
// Flash attention, causal, scale = sqrt(HS) = 8 (faithful to reference).
// qkv: [B*T, 3*E] bf16, q at col 0, k at col E, v at col 2E, head h at h*64.
// One block = one (b, h, 64-row query tile); 4 waves, each owns 16 rows.
// K tile is naturally [n=tk][k=d]; V tile is transposed on store to [n=d][k=s]
// so all B-fragments are contiguous ds_load_b128 pairs.
// ---------------------------------------------------------------------------
__global__ __launch_bounds__(128) void attention_kernel(
    const unsigned short* __restrict__ qkv, unsigned short* __restrict__ obf,
    int T) {
  __shared__ __align__(16) unsigned short Qs[64][72];
  __shared__ __align__(16) unsigned short Ks[64][72];  // [tk][d]
  __shared__ __align__(16) unsigned short Vt[64][72];  // [d][s]  (transposed)
  __shared__ __align__(16) unsigned short Ps[64][72];
  const int tid = threadIdx.x, lane = tid & 31, w = tid >> 5;
  const int qt = blockIdx.x, h = blockIdx.y, b = blockIdx.z;
  const int hf = lane >> 4, ln = lane & 15;

  {  // load Q tile
    int r = tid >> 1, c = (tid & 1) << 5;
    const uint4* src =
        (const uint4*)(qkv + (size_t)(b * T + qt * 64 + r) * 3072 + h * 64 + c);
    uint4* dst = (uint4*)&Qs[r][c];
#pragma unroll
    for (int j = 0; j < 4; ++j) dst[j] = src[j];
  }
  __syncthreads();
  v16bf aQ0 = load_frag_a(&Qs[0][0], 72, w * 16, 0, lane);
  v16bf aQ1 = load_frag_a(&Qs[0][0], 72, w * 16, 32, lane);

  float mrow[8], lrow[8];
  F32x8 oacc[4];
#pragma unroll
  for (int r = 0; r < 8; ++r) { mrow[r] = -1e30f; lrow[r] = 0.f; }
#pragma unroll
  for (int dt = 0; dt < 4; ++dt)
#pragma unroll
    for (int r = 0; r < 8; ++r) oacc[dt].f[r] = 0.f;

  for (int kt = 0; kt <= qt; ++kt) {
    {  // load K tile; load+transpose V tile
      int r = tid >> 1, c = (tid & 1) << 5;
      const uint4* ks =
          (const uint4*)(qkv + (size_t)(b * T + kt * 64 + r) * 3072 + 1024 + h * 64 + c);
      const uint4* vsrc =
          (const uint4*)(qkv + (size_t)(b * T + kt * 64 + r) * 3072 + 2048 + h * 64 + c);
      uint4* kd = (uint4*)&Ks[r][c];
#pragma unroll
      for (int j = 0; j < 4; ++j) kd[j] = ks[j];
#pragma unroll
      for (int j = 0; j < 4; ++j) {
        uint4 vv = vsrc[j];
        unsigned uu[4] = {vv.x, vv.y, vv.z, vv.w};
#pragma unroll
        for (int q = 0; q < 4; ++q) {
          int d0 = c + j * 8 + q * 2;
          Vt[d0][r]     = (unsigned short)(uu[q] & 0xffffu);
          Vt[d0 + 1][r] = (unsigned short)(uu[q] >> 16);
        }
      }
    }
    __syncthreads();

    // S = Q * K^T
    F32x8 s[4];
#pragma unroll
    for (int nt = 0; nt < 4; ++nt) {
#pragma unroll
      for (int r = 0; r < 8; ++r) s[nt].f[r] = 0.f;
      v16bf bK0 = load_frag_b_nk(&Ks[0][0], 72, nt * 16, 0, lane);
      s[nt].v = __builtin_amdgcn_wmma_f32_16x16x32_bf16(
          false, aQ0, false, bK0, (short)0, s[nt].v, false, false);
      v16bf bK1 = load_frag_b_nk(&Ks[0][0], 72, nt * 16, 32, lane);
      s[nt].v = __builtin_amdgcn_wmma_f32_16x16x32_bf16(
          false, aQ1, false, bK1, (short)0, s[nt].v, false, false);
    }

    // faithful scale (* sqrt(64) == 8) + causal mask
#pragma unroll
    for (int nt = 0; nt < 4; ++nt)
#pragma unroll
      for (int r = 0; r < 8; ++r) {
        int col = kt * 64 + nt * 16 + ln;
        int row = qt * 64 + w * 16 + r + 8 * hf;
        float sv = s[nt].f[r] * 8.0f;
        s[nt].f[r] = (col > row) ? -1e30f : sv;
      }

    // online softmax (row stats live in one 16-lane half-group)
#pragma unroll
    for (int r = 0; r < 8; ++r) {
      float rm = s[0].f[r];
#pragma unroll
      for (int nt = 1; nt < 4; ++nt) rm = fmaxf(rm, s[nt].f[r]);
      rm = fmaxf(rm, __shfl_xor(rm, 1));
      rm = fmaxf(rm, __shfl_xor(rm, 2));
      rm = fmaxf(rm, __shfl_xor(rm, 4));
      rm = fmaxf(rm, __shfl_xor(rm, 8));
      float mn = fmaxf(mrow[r], rm);
      float corr = __expf(mrow[r] - mn);
      float ps = 0.f;
      int prow = w * 16 + r + 8 * hf;
#pragma unroll
      for (int nt = 0; nt < 4; ++nt) {
        float p = __expf(s[nt].f[r] - mn);
        ps += p;
        Ps[prow][nt * 16 + ln] = f2bf(p);
      }
      ps += __shfl_xor(ps, 1);
      ps += __shfl_xor(ps, 2);
      ps += __shfl_xor(ps, 4);
      ps += __shfl_xor(ps, 8);
      lrow[r] = lrow[r] * corr + ps;
      mrow[r] = mn;
#pragma unroll
      for (int dt = 0; dt < 4; ++dt) oacc[dt].f[r] *= corr;
    }

    // O += P * V (Ps rows wave-private; per-wave DS ops are in-order)
    v16bf aP0 = load_frag_a(&Ps[0][0], 72, w * 16, 0, lane);
    v16bf aP1 = load_frag_a(&Ps[0][0], 72, w * 16, 32, lane);
#pragma unroll
    for (int dt = 0; dt < 4; ++dt) {
      v16bf bV0 = load_frag_b_nk(&Vt[0][0], 72, dt * 16, 0, lane);
      oacc[dt].v = __builtin_amdgcn_wmma_f32_16x16x32_bf16(
          false, aP0, false, bV0, (short)0, oacc[dt].v, false, false);
      v16bf bV1 = load_frag_b_nk(&Vt[0][0], 72, dt * 16, 32, lane);
      oacc[dt].v = __builtin_amdgcn_wmma_f32_16x16x32_bf16(
          false, aP1, false, bV1, (short)0, oacc[dt].v, false, false);
    }
    __syncthreads();  // protect Ks/Vt before next iteration's overwrite
  }

  // write O (head-concat layout [B*T, H*HS])
#pragma unroll
  for (int dt = 0; dt < 4; ++dt)
#pragma unroll
    for (int r = 0; r < 8; ++r) {
      int row = b * T + qt * 64 + w * 16 + r + 8 * hf;
      int col = h * 64 + dt * 16 + ln;
      obf[(size_t)row * 1024 + col] = f2bf(oacc[dt].f[r] / lrow[r]);
    }
}

// ---------------------------------------------------------------------------
// LayerNorm (unbiased var, ddof=1, eps=1e-5) -> bf16. One block per row.
// ---------------------------------------------------------------------------
__global__ __launch_bounds__(256) void ln_kernel(
    const float* __restrict__ x, const float* __restrict__ g,
    const float* __restrict__ be, unsigned short* __restrict__ out) {
  __shared__ float red[256];
  const int tid = threadIdx.x;
  const size_t row = blockIdx.x;
  const float* xr = x + row * 1024;
  float v[4];
  float s = 0.f;
#pragma unroll
  for (int j = 0; j < 4; ++j) { v[j] = xr[tid + j * 256]; s += v[j]; }
  red[tid] = s; __syncthreads();
  for (int st = 128; st > 0; st >>= 1) {
    if (tid < st) red[tid] += red[tid + st];
    __syncthreads();
  }
  float mean = red[0] * (1.0f / 1024.0f);
  __syncthreads();
  float s2 = 0.f;
#pragma unroll
  for (int j = 0; j < 4; ++j) { float d = v[j] - mean; s2 += d * d; }
  red[tid] = s2; __syncthreads();
  for (int st = 128; st > 0; st >>= 1) {
    if (tid < st) red[tid] += red[tid + st];
    __syncthreads();
  }
  float inv = rsqrtf(red[0] * (1.0f / 1023.0f) + 1e-5f);
#pragma unroll
  for (int j = 0; j < 4; ++j) {
    int c = tid + j * 256;
    out[row * 1024 + c] = f2bf((v[j] - mean) * inv * g[c] + be[c]);
  }
}

// ---------------------------------------------------------------------------
// Weight conversions to bf16, TRANSPOSED to [N][K] for the GEMM B operand.
// ---------------------------------------------------------------------------
// in: [K][N] f32 row-major -> out: [N][K] bf16 row-major
__global__ void cvt_t_kernel(const float* __restrict__ in,
                             unsigned short* __restrict__ out, int K, int N) {
  long i = (long)blockIdx.x * blockDim.x + threadIdx.x;
  if (i >= (long)K * N) return;
  int n = (int)(i / K);
  int k = (int)(i % K);
  out[i] = f2bf(in[(size_t)k * N + n]);
}

// Wq/Wk/Wv [H=16, E=1024, HS=64] -> [N=3072][K=1024]: row n = head-major q|k|v.
__global__ void cvt_qkv_kernel(const float* __restrict__ Wq,
                               const float* __restrict__ Wk,
                               const float* __restrict__ Wv,
                               unsigned short* __restrict__ out) {
  long i = (long)blockIdx.x * 256 + threadIdx.x;  // 3072 * 1024 total
  int n = (int)(i / 1024);
  int e = (int)(i % 1024);
  const float* W = (n < 1024) ? Wq : ((n < 2048) ? Wk : Wv);
  int nn = n & 1023;
  int h = nn >> 6, d = nn & 63;
  out[i] = f2bf(W[((size_t)h * 1024 + e) * 64 + d]);
}

// ---------------------------------------------------------------------------
extern "C" void kernel_launch(void* const* d_in, const int* in_sizes, int n_in,
                              void* d_out, int out_size, void* d_ws, size_t ws_size,
                              hipStream_t stream) {
  const float* x   = (const float*)d_in[0];
  const float* Wq  = (const float*)d_in[1];
  const float* Wk  = (const float*)d_in[2];
  const float* Wv  = (const float*)d_in[3];
  const float* Wp  = (const float*)d_in[4];
  const float* bp  = (const float*)d_in[5];
  const float* W1  = (const float*)d_in[6];
  const float* b1  = (const float*)d_in[7];
  const float* W2  = (const float*)d_in[8];
  const float* b2  = (const float*)d_in[9];
  const float* g1  = (const float*)d_in[10];
  const float* be1 = (const float*)d_in[11];
  const float* g2  = (const float*)d_in[12];
  const float* be2 = (const float*)d_in[13];
  (void)in_sizes; (void)n_in; (void)out_size; (void)ws_size;

  const int Bv = 4, T = 2048, E = 1024, Hh = 16;
  const int M = Bv * T;  // 8192

  char* ws = (char*)d_ws;
  size_t off = 0;
  auto walloc = [&](size_t bytes) -> void* {
    void* p = (void*)(ws + off);
    off = (off + bytes + 255) & ~(size_t)255;
    return p;
  };
  unsigned short* hbf  = (unsigned short*)walloc((size_t)M * E * 2);      // LN1 out
  unsigned short* wqkv = (unsigned short*)walloc((size_t)3 * E * E * 2);  // [3E][E]
  unsigned short* qkv  = (unsigned short*)walloc((size_t)M * 3 * E * 2);  // q|k|v
  unsigned short* obf  = (unsigned short*)walloc((size_t)M * E * 2);      // attn out
  unsigned short* wpb  = (unsigned short*)walloc((size_t)E * E * 2);      // [E][E]^T
  float*          x1   = (float*)walloc((size_t)M * E * 4);               // x + proj
  unsigned short* h2   = (unsigned short*)walloc((size_t)M * E * 2);      // LN2 out
  unsigned short* w1b  = (unsigned short*)walloc((size_t)4 * E * E * 2);  // [4E][E]
  unsigned short* w2b  = (unsigned short*)walloc((size_t)4 * E * E * 2);  // [E][4E]
  unsigned short* f1   = (unsigned short*)walloc((size_t)M * 4 * E * 2);  // relu(ffn1)

  // weight conversions (transposed B operands), every call: deterministic
  cvt_qkv_kernel<<<(3L * E * E) / 256, 256, 0, stream>>>(Wq, Wk, Wv, wqkv);
  cvt_t_kernel<<<((long)E * E + 255) / 256, 256, 0, stream>>>(Wp, wpb, E, E);
  cvt_t_kernel<<<((long)4 * E * E + 255) / 256, 256, 0, stream>>>(W1, w1b, E, 4 * E);
  cvt_t_kernel<<<((long)4 * E * E + 255) / 256, 256, 0, stream>>>(W2, w2b, 4 * E, E);

  // h = LN1(x)
  ln_kernel<<<M, 256, 0, stream>>>(x, g1, be1, hbf);
  // qkv = h @ Wqkv
  gemm_bf16_kernel<<<dim3(3 * E / 256, M / 64), 256, 0, stream>>>(
      hbf, wqkv, nullptr, nullptr, nullptr, qkv, M, 3 * E, E, 0);
  // o = attention(q,k,v)
  attention_kernel<<<dim3(T / 64, Hh, Bv), 128, 0, stream>>>(qkv, obf, T);
  // x1 = x + (o @ Wp + bp)
  gemm_bf16_kernel<<<dim3(E / 256, M / 64), 256, 0, stream>>>(
      obf, wpb, bp, x, x1, nullptr, M, E, E, 0);
  // h2 = LN2(x1)
  ln_kernel<<<M, 256, 0, stream>>>(x1, g2, be2, h2);
  // f1 = relu(h2 @ W1 + b1)
  gemm_bf16_kernel<<<dim3(4 * E / 256, M / 64), 256, 0, stream>>>(
      h2, w1b, b1, nullptr, nullptr, f1, M, 4 * E, E, 1);
  // out = x1 + (f1 @ W2 + b2)
  gemm_bf16_kernel<<<dim3(E / 256, M / 64), 256, 0, stream>>>(
      f1, w2b, b2, x1, (float*)d_out, nullptr, M, E, 4 * E, 0);
}
